// KarcherFFN_35253091565662
// MI455X (gfx1250) — matrix operand order repaired and checked by reading, not compiled, not run on hardware
//
#include <hip/hip_runtime.h>
#include <hip/hip_bf16.h>
#include <cmath>

// ---------------------------------------------------------------------------
// KarcherFFN on MI455X (gfx1250, wave32, WMMA)
//
// Per-token: routing softmax -> wr[32]; TT cores = basis^T[6144x32] x wr
// (WMMA, operandswapped: A=basis tile, B=wr -> C rows are core elements, so
// each lane stores contiguous halves); TT1 256->64 (VALU, reg-resident cA1
// rows); TT2 temp (VALU); TT2 output 32x64x32 (WMMA); exact gelu in C regs;
// down-proj 1024->256 (WMMA, 32 chained K-steps).
//
// ~18MB HBM traffic (~0.8us @ 23.3TB/s) vs ~9.3GFLOP -> compute bound; >80%
// of MACs on v_wmma_f32_16x16x32_f16; VALU ~5K FMA/lane per 16-token tile.
//
// ws layout (~1.5MB):
//   [0)        basis A-fragments  f16 (384 tiles * 32 lanes * 16 halves)
//   [393216)   w_down B-fragments f16 (512 tiles * 32 lanes * 16 halves)
//   [917504)   wr f16 [8192][32]  (WMMA B rows: lane reads [tok][hi*16..+15])
// ---------------------------------------------------------------------------

typedef __attribute__((ext_vector_type(16))) _Float16 v16h;
typedef __attribute__((ext_vector_type(8)))  _Float16 v8h;
typedef __attribute__((ext_vector_type(8)))  float    v8f;

#define WS_BF_OFF  0
#define WS_WD_OFF  393216
#define WS_WR_OFF  917504

#define NBF (384*32*16)
#define NWD (512*32*16)

union AFrag { v16h v; v8h h[2]; };

// A-fragment from a row-major [M][rowlen] f16 buffer: lane L holds row M=L&15,
// halves {koff..koff+15}, koff=(L>=16)?16:0. Pass (row base + koff).
static __device__ inline v16h load_afrag(const _Float16* p) {
  AFrag u;
  u.h[0] = *(const v8h*)(p);
  u.h[1] = *(const v8h*)(p + 8);
  return u.v;
}

static __device__ inline float gelu_erf(float v) {
  return 0.5f * v * (1.f + erff(v * 0.70710678118f));
}

// ---------------------------------------------------------------------------
// Kernel 1: pre-swizzle.
//  - basis banks -> A-fragment order: lane L row M=L&15 (core element within
//    tile), K halves contiguous: K = 2*koff + e = hi*16 + e.
//  - w_down -> B-fragment order: lane L col N=L&15, K = hi*16 + e.
// ---------------------------------------------------------------------------
__global__ void k_preswizzle(const float* __restrict__ A1, const float* __restrict__ A2,
                             const float* __restrict__ B1, const float* __restrict__ B2,
                             const float* __restrict__ Wd,
                             _Float16* __restrict__ bf, _Float16* __restrict__ wd) {
  int g = blockIdx.x * blockDim.x + threadIdx.x;
  if (g < NBF) {
    int e    = g & 15;
    int lane = (g >> 4) & 31;
    int ct   = g >> 9;                        // tile 0..383 over m in [0,6144)
    int koff = (lane >> 4) * 8;
    int n    = 2*koff + e;   // basis K
    int m    = ct * 16 + (lane & 15);                          // core element
    float v;
    if      (m < 1024) v = A1[n*1024 +  m        ];   // [n,i16,r8,k8]
    else if (m < 2048) v = A2[n*1024 + (m - 1024)];   // [n,r8,j16,l8]
    else if (m < 4096) v = B1[n*2048 + (m - 2048)];   // [n,i8,r8,k32]
    else               v = B2[n*2048 + (m - 4096)];   // [n,r8,j8,l32]
    bf[g] = (_Float16)v;
  } else if (g < NBF + NWD) {
    int gw   = g - NBF;
    int e    = gw & 15;
    int lane = (gw >> 4) & 31;
    int tile = gw >> 9;                       // tile = kt*16 + nt
    int kt   = tile >> 4, nt = tile & 15;
    int k    = kt*32 + ((lane >> 4) << 4) + e;
    int c    = nt*16 + (lane & 15);
    wd[gw] = (_Float16)Wd[k*256 + c];
  }
}

// ---------------------------------------------------------------------------
// Kernel 2: routing.  One wave per token; lane = basis index n.
// ---------------------------------------------------------------------------
__global__ void k_routing(const int* __restrict__ nidx, const float* __restrict__ nw,
                          const float* __restrict__ recipes,
                          _Float16* __restrict__ wrout) {
  int lane  = threadIdx.x & 31;
  int token = blockIdx.x * (blockDim.x >> 5) + (threadIdx.x >> 5);
  float acc = 0.f;
  for (int k = 0; k < 8; ++k) {
    int   id = nidx[token*8 + k];
    float r  = recipes[id*32 + lane];
    float m  = r;
    for (int off = 16; off; off >>= 1) m = fmaxf(m, __shfl_xor(m, off, 32));
    float e = __expf(r - m);
    float s = e;
    for (int off = 16; off; off >>= 1) s += __shfl_xor(s, off, 32);
    acc += (e / s) * nw[token*8 + k];
  }
  wrout[token*32 + lane] = (_Float16)acc;
}

// ---------------------------------------------------------------------------
// Kernel 3: fused 16-token tile.  128 threads = 4 waves.  Dynamic LDS 200704B:
//   [0)       cores: per token 2048 halves.
//             stage1: cA1 as [k8][ir128] (k*128+ir), cA2 linear at +1024
//             stage2: cB1 as [k32][ir64] (k*64+ir)
//   [65536)   cb2f:  cB2 in B-frag order, K-order kappa=j*8+r
//   [131072)  t2buf: T2[k16][kappa64] f16 per token, 32KB (overlaps xT 16KB)
//   [163840)  act:   16x1024 f16
//   [196608)  hbuf:  16x64 f32
// ---------------------------------------------------------------------------
__global__ __launch_bounds__(128, 1)
void k_ttffn(const float* __restrict__ x, const _Float16* __restrict__ wr,
             const _Float16* __restrict__ bf, const _Float16* __restrict__ wdf,
             const float* __restrict__ bias, float* __restrict__ out) {
  extern __shared__ __align__(16) char smem[];
  _Float16* cores = (_Float16*)smem;                      // 65536 B
  _Float16* cb2f  = (_Float16*)(smem + 65536);            // 65536 B
  _Float16* t2buf = (_Float16*)(smem + 131072);           // 32768 B
  float*    ldsx  = (float*)   (smem + 131072);           // 16384 B (xT, dead after TT1)
  _Float16* act   = (_Float16*)(smem + 163840);           // 32768 B
  float*    hbuf  = (float*)   (smem + 196608);           // 4096 B

  const int tid  = threadIdx.x;
  const int lane = tid & 31, wave = tid >> 5;
  const int hi   = lane >> 4, col = lane & 15;
  const int tb   = blockIdx.x * 16;

  // ---- P0: stage x tile transposed: ldsx[t][j*16+i] = x[t][i*16+j] ----
  for (int q = 0; q < 8; ++q) {
    int fi = q*128 + tid;
    int t  = fi >> 6, w4 = fi & 63;
    float4 xv = ((const float4*)(x + (size_t)(tb + t)*256))[w4];
    int c0 = w4*4;
    int i = c0 >> 4, j0 = c0 & 15;           // 4 consecutive j, same i
    float* dst = ldsx + t*256 + j0*16 + i;
    dst[0] = xv.x; dst[16] = xv.y; dst[32] = xv.z; dst[48] = xv.w;
  }
  // Loop-invariant B-fragment: wr column block for this tile's 16 tokens.
  // B layout: lane L -> col N = token col, K = hi*16 + e  (contiguous 32B).
  v16h b_wr = *(const v16h*)(wr + (size_t)(tb + col)*32 + hi*16);
  __syncthreads();

  // ---- P1: WMMA core synthesis stage 1. C: M=core elem (v+8hi), N=token(col)
  // cA1 tiles 0..63 -> [k][ir] (ir = i*8+r): m = ir*8+k, group has k=v. ----
  for (int ct = wave; ct < 64; ct += 4) {
    v16h a = *(const v16h*)(bf + (size_t)(ct*32 + lane)*16);
    v8f  c = {};
    c = __builtin_amdgcn_wmma_f32_16x16x32_f16(false, a, false, b_wr,
                                               (short)0, c, false, false);
    _Float16* dst = cores + col*2048 + (ct*2 + hi);    // + k*128
#pragma unroll
    for (int v = 0; v < 8; ++v) dst[v*128] = (_Float16)c[v];
  }
  // cA2 tiles 64..127 -> linear at +1024: 8 consecutive halves -> one b128.
  for (int ct = 64 + wave; ct < 128; ct += 4) {
    v16h a = *(const v16h*)(bf + (size_t)(ct*32 + lane)*16);
    v8f  c = {};
    c = __builtin_amdgcn_wmma_f32_16x16x32_f16(false, a, false, b_wr,
                                               (short)0, c, false, false);
    v8h p;
#pragma unroll
    for (int v = 0; v < 8; ++v) p[v] = (_Float16)c[v];
    *(v8h*)(cores + col*2048 + 1024 + (ct - 64)*16 + 8*hi) = p;
  }
  __syncthreads();

  // ---- P2: TT1 (256->64), VALU.  8 lanes/token, lane owns k.  cA1 k-row is
  //      contiguous -> preload 16 v8h into registers, reuse across all j. ----
  {
    int t = tid >> 3, k = tid & 7;
    const v8h* A1p = (const v8h*)(cores + t*2048 + k*128);   // [i] -> r halves
    v8h A1r[16];
#pragma unroll
    for (int i = 0; i < 16; ++i) A1r[i] = A1p[i];
    const v8h* A2p = (const v8h*)(cores + t*2048 + 1024);    // [(r*16+j)] -> l
    const float* xt = ldsx + t*256;
    float hacc[8];
#pragma unroll
    for (int l = 0; l < 8; ++l) hacc[l] = 0.f;
    for (int j = 0; j < 16; ++j) {
      float t1[8];
#pragma unroll
      for (int r = 0; r < 8; ++r) t1[r] = 0.f;
      const float4* xr = (const float4*)(xt + j*16);
#pragma unroll
      for (int i4 = 0; i4 < 4; ++i4) {
        float4 xv = xr[i4];
        float xs[4] = {xv.x, xv.y, xv.z, xv.w};
#pragma unroll
        for (int ii = 0; ii < 4; ++ii)
#pragma unroll
          for (int r = 0; r < 8; ++r) t1[r] += xs[ii] * (float)A1r[i4*4 + ii][r];
      }
#pragma unroll
      for (int r = 0; r < 8; ++r) {
        v8h c2 = A2p[r*16 + j];
#pragma unroll
        for (int l = 0; l < 8; ++l) hacc[l] += t1[r] * (float)c2[l];
      }
    }
#pragma unroll
    for (int l = 0; l < 8; ++l) hbuf[t*64 + k*8 + l] = hacc[l];
  }
  __syncthreads();

  // ---- P3a: cB1 tiles 128..255 -> [k][ir] (ir=i*8+r): m = ir*32+k ----
  for (int ct = 128 + wave; ct < 256; ct += 4) {
    v16h a = *(const v16h*)(bf + (size_t)(ct*32 + lane)*16);
    v8f  c = {};
    c = __builtin_amdgcn_wmma_f32_16x16x32_f16(false, a, false, b_wr,
                                               (short)0, c, false, false);
    int m0 = (ct - 128)*16 + 8*hi;
    int k0 = m0 & 31, ir0 = m0 >> 5;          // group: fixed ir, k = k0+v
    _Float16* dst = cores + col*2048 + k0*64 + ir0;    // + v*64
#pragma unroll
    for (int v = 0; v < 8; ++v) dst[v*64] = (_Float16)c[v];
  }
  // ---- P3b: cB2 tiles 256..383 -> B-frag order with K-order kappa=j*8+r ----
  for (int ct = 256 + wave; ct < 384; ct += 4) {
    v16h a = *(const v16h*)(bf + (size_t)(ct*32 + lane)*16);
    v8f  c = {};
    c = __builtin_amdgcn_wmma_f32_16x16x32_f16(false, a, false, b_wr,
                                               (short)0, c, false, false);
    int m0    = (ct - 256)*16 + 8*hi;         // (r*8+j)*32 + l, group: l=l0+v
    int rjraw = m0 >> 5;
    int kap   = (rjraw & 7)*8 + (rjraw >> 3); // kappa = j*8 + r
    int ks    = kap >> 5;
    int hi2   = (kap >> 4) & 1;
    int e     = kap & 15;
    int l0    = m0 & 31;
    int nt    = l0 >> 4;
    _Float16* dst = cb2f + col*2048 + (ks*2 + nt)*512 + (hi2*16 + (l0 & 15))*16 + e;
#pragma unroll
    for (int v = 0; v < 8; ++v) dst[v*16] = (_Float16)c[v];   // + l step
  }
  __syncthreads();

  // ---- P4: TT2 in two k-halves: VALU temp2 -> T2 f16, then WMMA 32x64x32 ----
  for (int mt = 0; mt < 2; ++mt) {
    // P4a: T2[k][j*8+r] = sum_i h[i*8+j]*cB1[k][i*8+r]; 2 k-rows per thread.
    {
      int t = tid >> 3, sub = tid & 7;
      int kA = mt*16 + sub*2;
      const v8h* B1p = (const v8h*)(cores + t*2048);   // [k*8 + i] -> r halves
      v8h R0[8], R1[8];
#pragma unroll
      for (int i = 0; i < 8; ++i) { R0[i] = B1p[(kA+0)*8 + i]; R1[i] = B1p[(kA+1)*8 + i]; }
      const float* ht  = hbuf + t*64;
      _Float16*    t2r = t2buf + t*1024;
      for (int j = 0; j < 8; ++j) {
        float ta[8], tc[8];
#pragma unroll
        for (int r = 0; r < 8; ++r) { ta[r] = 0.f; tc[r] = 0.f; }
#pragma unroll
        for (int i = 0; i < 8; ++i) {
          float hv = ht[i*8 + j];
#pragma unroll
          for (int r = 0; r < 8; ++r) {
            ta[r] += hv * (float)R0[i][r];
            tc[r] += hv * (float)R1[i][r];
          }
        }
        v8h pa, pc;
#pragma unroll
        for (int r = 0; r < 8; ++r) { pa[r] = (_Float16)ta[r]; pc[r] = (_Float16)tc[r]; }
        *(v8h*)(t2r + (sub*2 + 0)*64 + j*8) = pa;
        *(v8h*)(t2r + (sub*2 + 1)*64 + j*8) = pc;
      }
    }
    __syncthreads();
    // P4b: out[k,l] = T2[16x64] x cB2[64x32] per token; gelu in C regs -> act.
    for (int job = wave; job < 32; job += 4) {     // 16 tokens x 2 N-tiles
      int t = job >> 1, nt = job & 1;
      v8f acc = {};
#pragma unroll
      for (int ks = 0; ks < 2; ++ks) {
        v16h a = load_afrag(t2buf + t*1024 + col*64 + ks*32 + hi*16);
        v16h b = *(const v16h*)(cb2f + t*2048 + (ks*2 + nt)*512 + lane*16);
        acc = __builtin_amdgcn_wmma_f32_16x16x32_f16(false, a, false, b,
                                                     (short)0, acc, false, false);
      }
#pragma unroll
      for (int v = 0; v < 8; ++v) {                // hidden = k*32 + l
        int k = mt*16 + v + 8*hi;
        int l = nt*16 + col;
        act[t*1024 + k*32 + l] = (_Float16)gelu_erf(acc[v]);
      }
    }
    __syncthreads();
  }

  // ---- P5: WMMA down-proj [16x1024]x[1024x256], 32 chained K steps ----
  {
    int nt0 = wave * 4;
    v8f acc[4];
#pragma unroll
    for (int q = 0; q < 4; ++q) acc[q] = (v8f){};
    for (int kt = 0; kt < 32; ++kt) {
      v16h a = load_afrag(act + col*1024 + kt*32 + hi*16);
#pragma unroll
      for (int q = 0; q < 4; ++q) {
        int  nt = nt0 + q;
        v16h b  = *(const v16h*)(wdf + (size_t)((kt*16 + nt)*32 + lane)*16);
        acc[q] = __builtin_amdgcn_wmma_f32_16x16x32_f16(false, a, false, b,
                                                        (short)0, acc[q],
                                                        false, false);
      }
    }
#pragma unroll
    for (int q = 0; q < 4; ++q) {
      int   cidx = (nt0 + q)*16 + col;
      float bv   = bias[cidx];
#pragma unroll
      for (int v = 0; v < 8; ++v)
        out[(size_t)(tb + v + 8*hi)*256 + cidx] = acc[q][v] + bv;
    }
  }
}

// ---------------------------------------------------------------------------
extern "C" void kernel_launch(void* const* d_in, const int* in_sizes, int n_in,
                              void* d_out, int out_size, void* d_ws, size_t ws_size,
                              hipStream_t stream) {
  const float* x       = (const float*)d_in[0];
  const int*   nidx    = (const int*)  d_in[1];
  const float* nw      = (const float*)d_in[2];
  const float* recipes = (const float*)d_in[3];
  const float* A1      = (const float*)d_in[4];
  const float* A2      = (const float*)d_in[5];
  const float* B1      = (const float*)d_in[6];
  const float* B2      = (const float*)d_in[7];
  const float* Wd      = (const float*)d_in[8];
  const float* bias    = (const float*)d_in[9];
  float*       out     = (float*)d_out;

  char* ws = (char*)d_ws;
  _Float16* bf  = (_Float16*)(ws + WS_BF_OFF);
  _Float16* wd  = (_Float16*)(ws + WS_WD_OFF);
  _Float16* wrp = (_Float16*)(ws + WS_WR_OFF);

  k_preswizzle<<<(NBF + NWD + 255) / 256, 256, 0, stream>>>(A1, A2, B1, B2, Wd, bf, wd);
  k_routing<<<2048, 128, 0, stream>>>(nidx, nw, recipes, wrp);

  const size_t shmem = 200704;
  (void)hipFuncSetAttribute((const void*)k_ttffn,
                            hipFuncAttributeMaxDynamicSharedMemorySize, (int)shmem);
  k_ttffn<<<512, 128, shmem, stream>>>(x, wrp, bf, wd, bias, out);
}
